// ResidualScaledDotProductAttention_41472204210593
// MI455X (gfx1250) — matrix operand order repaired and checked by reading, not compiled, not run on hardware
//
#include <hip/hip_runtime.h>

// Problem constants (reference: B=4, Sq=Sk=2048, D=512, fp32)
constexpr int NB  = 4;
constexpr int NSQ = 2048;
constexpr int NSK = 2048;
constexpr int ND  = 512;
constexpr int TBM = 32;   // q rows per workgroup
constexpr int TBN = 64;   // keys per k-tile
constexpr int NKT = NSK / TBN;
constexpr float SCALE = 0.044194173824159216f; // 1/sqrt(512)
constexpr float PEN   = 10000.0f;

// padded LDS strides (bf16 elements) -> conflict-free ds_read_b128
constexpr int QS_LD = ND + 8;    // 520
constexpr int KV_LD = 64 + 8;    // 72  (144B row stride == TDM pad: 128B row + 16B pad)
constexpr int PS_LD = 64 + 8;    // 72

typedef __attribute__((ext_vector_type(16))) __bf16 v16bf;
typedef __attribute__((ext_vector_type(8)))  float  v8f;
typedef __attribute__((ext_vector_type(4)))  unsigned su4;
typedef __attribute__((ext_vector_type(8)))  unsigned su8;

// ---------------- bf16 conversion helpers (pre-pass only) ----------------
__device__ __forceinline__ unsigned short f2bf(float f) {
  union { float f; unsigned u; } x; x.f = f;
  unsigned r = x.u + 0x7FFFu + ((x.u >> 16) & 1u);   // round-to-nearest-even
  return (unsigned short)(r >> 16);
}
__device__ __forceinline__ void cvt2(unsigned short* dst, float a, float b) {
#if __has_builtin(__builtin_amdgcn_cvt_pk_bf16_f32)
  auto r = __builtin_amdgcn_cvt_pk_bf16_f32(a, b);   // v_cvt_pk_bf16_f32
  __builtin_memcpy(dst, &r, 4);
#else
  dst[0] = f2bf(a); dst[1] = f2bf(b);
#endif
}

// ---------------- CDNA5 async global->LDS copy (per-lane, raw bytes) ----------------
__device__ __forceinline__ void async_b128(unsigned lds, const void* gptr) {
  asm volatile("global_load_async_to_lds_b128 %0, %1, off"
               :: "v"(lds), "v"((unsigned long long)(size_t)gptr) : "memory");
}
__device__ __forceinline__ void async_wait0() {
  asm volatile("s_wait_asynccnt 0x0" ::: "memory");
}

// ---------------- CDNA5 Tensor Data Mover: one 64x64 bf16 tile per instruction ----
// D# group0: count=1 | lds_addr | global_addr | type=2 ("image")
// D# group1: data_size=2B, pad_enable, pad_interval=4 (pad after 32 DWORDs = 128B row),
//            pad_amount=3 (4 DWORDs = 16B)  -> LDS row stride 144B = KV_LD*2
__device__ __forceinline__ void tdm_load_tile64(unsigned lds_byte, unsigned long long gaddr,
                                                unsigned row_stride_elems) {
  su4 g0;
  g0[0] = 1u;                                            // count=1, user descriptor
  g0[1] = lds_byte;                                      // lds_addr
  g0[2] = (unsigned)gaddr;                               // global_addr[31:0]
  g0[3] = ((unsigned)(gaddr >> 32) & 0x01FFFFFFu) | (2u << 30); // addr[56:32] | type=2
  su8 g1;
  g1[0] = (1u << 16) | (1u << 20) | (4u << 22) | (3u << 25);    // 2B | pad_en | intv | amt
  g1[1] = 0u;                   // abar=0; tensor_dim0[15:0] (dim0=1<<20 -> low16 0)
  g1[2] = (1u << 4);            // tensor_dim0[31:16]=0x10 ; tensor_dim1[15:0]=0
  g1[3] = (1u << 4) | (64u << 16); // tensor_dim1[31:16]=0x10 ; tile_dim0 = 64
  g1[4] = 64u;                  // tile_dim1 = 64 ; tile_dim2 = 0
  g1[5] = row_stride_elems;     // tensor_dim0_stride[31:0] (data_size units)
  g1[6] = 0u;                   // stride[47:32]=0 ; tensor_dim1_stride[15:0]=0
  g1[7] = 0u;
  asm volatile("tensor_load_to_lds %0, %1" :: "s"(g0), "s"(g1) : "memory");
}
__device__ __forceinline__ void tdm_wait0() {
  __builtin_amdgcn_s_wait_tensorcnt(0);   // s_wait_tensorcnt 0 (codegen-confirmed builtin)
}

__device__ __forceinline__ v16bf ld_frag(const unsigned short* s, int off0, int off1) {
  union { v16bf v; float4 f[2]; } u;
  u.f[0] = *reinterpret_cast<const float4*>(s + off0);
  u.f[1] = *reinterpret_cast<const float4*>(s + off1);
  return u.v;
}

// ================= pre-pass 1: fp32 -> bf16, layout preserved =================
__global__ __launch_bounds__(256)
void cvt_bf16_kernel(const float* __restrict__ src, unsigned short* __restrict__ dst) {
  int i = blockIdx.x * 256 + threadIdx.x;            // float4 index
  #pragma unroll
  for (int rep = 0; rep < 2; ++rep, i += 524288) {   // 2048 blocks cover 4 Mi elements
    float4 v = reinterpret_cast<const float4*>(src)[i];
    unsigned short* d = dst + (size_t)i * 4;
    cvt2(d, v.x, v.y);
    cvt2(d + 2, v.z, v.w);
  }
}

// ========= pre-pass 2: V fp32 [B][Sk][D] -> bf16 transposed Vg [B][D][Sk] =========
__global__ __launch_bounds__(256)
void transpose_v_kernel(const float* __restrict__ V, unsigned short* __restrict__ Vg) {
  __shared__ __align__(16) unsigned short T[64 * KV_LD];
  const int tid = threadIdx.x;
  const int kt = blockIdx.x, dt = blockIdx.y, b = blockIdx.z;
  const long base = ((long)b * NSK + kt * 64) * ND + dt * 64;
  #pragma unroll
  for (int it = 0; it < 4; ++it) {                   // read 64 keys x 64 d, coalesced
    int i = tid + it * 256;
    int r = i >> 4, c4 = (i & 15) << 2;
    float4 v = *reinterpret_cast<const float4*>(V + base + (long)r * ND + c4);
    T[(c4 + 0) * KV_LD + r] = f2bf(v.x);
    T[(c4 + 1) * KV_LD + r] = f2bf(v.y);
    T[(c4 + 2) * KV_LD + r] = f2bf(v.z);
    T[(c4 + 3) * KV_LD + r] = f2bf(v.w);
  }
  __syncthreads();
  const long obase = ((long)b * ND + dt * 64) * NSK + kt * 64;
  #pragma unroll
  for (int it = 0; it < 2; ++it) {                   // write d-major rows, coalesced b128
    int i = tid + it * 256;
    int r = i >> 3, s = i & 7;
    *reinterpret_cast<float4*>(Vg + obase + (long)r * NSK + s * 8) =
        *reinterpret_cast<const float4*>(&T[r * KV_LD + s * 8]);
  }
}

// ======================= fused attention main kernel =======================
__global__ __launch_bounds__(256)
void fa_resid_kernel(const unsigned short* __restrict__ Qb,
                     const unsigned short* __restrict__ Kb,
                     const unsigned short* __restrict__ Vg,
                     const float* __restrict__ Pv,
                     const unsigned char* __restrict__ Mk,
                     float* __restrict__ Ov, float* __restrict__ Np)
{
  __shared__ __align__(16) unsigned short Qs[TBM * QS_LD];    // 32x512 bf16 Q tile
  __shared__ __align__(16) unsigned short KV2[2][64 * KV_LD]; // double-buffered K / V^T tiles
  __shared__ __align__(16) unsigned short Ps[TBM * PS_LD];    // 32x64 bf16 probabilities
  __shared__ float s_red[2 * 4 * TBM];

  const int tid  = threadIdx.x;
  const int lane = tid & 31;
  const int w    = tid >> 5;
  const int lm   = lane & 15;
  const int hl   = lane >> 4;
  const int rh   = w & 1;        // q-row half
  const int cq   = w >> 1;       // column quarter

  const int q0 = blockIdx.x * TBM;
  const int b  = blockIdx.y;
  const long bq = (long)b * NSQ;
  const long bk = (long)b * NSK;

  const unsigned qs_b  = (unsigned)(size_t)&Qs[0];
  const unsigned kv_b[2] = { (unsigned)(size_t)&KV2[0][0], (unsigned)(size_t)&KV2[1][0] };

  // ---- per-lane async copy of the whole Q tile (bf16, already in layout)
  #pragma unroll
  for (int it = 0; it < 8; ++it) {
    int i = tid + it * 256;
    int r = i >> 6, s = i & 63;
    async_b128(qs_b + r * (QS_LD * 2) + s * 16, Qb + (bq + q0 + r) * ND + s * 8);
  }
  // ---- TDM: first K chunk (kt=0, d-chunk 0) into buffer 0 (one DMA for the 64x64 tile)
  if (w == 0) tdm_load_tile64(kv_b[0], (unsigned long long)(size_t)(Kb + bk * ND), ND);

  float m_i[8], l_i[8], corr[8], sv[8], tm[8];
  v8f o_acc[8];
  const v8f vzero = {0.f,0.f,0.f,0.f,0.f,0.f,0.f,0.f};
  #pragma unroll
  for (int t = 0; t < 8; ++t) { m_i[t] = -1e30f; l_i[t] = 0.f; o_acc[t] = vzero; }

  async_wait0();                  // this wave's Q-tile pieces have landed

  const unsigned char* mrow = Mk + bk;

  for (int kt = 0; kt < NKT; ++kt) {
    const int k0 = kt * TBN;
    const bool active = (k0 <= q0 + TBM - 1);   // uniform across block

    // ============ S = Q K^T : 8 pipelined 64-wide d-chunks ============
    v8f sacc = vzero;
    for (int c = 0; c < 8; ++c) {
      if (w == 0) tdm_wait0();    // issuing wave's in-flight tile has landed
      __syncthreads();            // publish tile; previous buffer fully consumed
      if (w == 0) {               // prefetch next tile via TDM into the other buffer
        if (c < 7) {
          tdm_load_tile64(kv_b[(c + 1) & 1],
                          (unsigned long long)(size_t)(Kb + (bk + k0) * ND + (c + 1) * 64), ND);
        } else if (active) {      // first V^T chunk for the PV phase
          tdm_load_tile64(kv_b[0],
                          (unsigned long long)(size_t)(Vg + ((long)b * ND) * NSK + k0), NSK);
        } else if (kt + 1 < NKT) {// masked tile: next k-tile's first K chunk
          tdm_load_tile64(kv_b[0],
                          (unsigned long long)(size_t)(Kb + (bk + k0 + TBN) * ND), ND);
        }
      }
      const unsigned short* KsC = &KV2[c & 1][0];
      const int dc = c * 64;
      #pragma unroll
      for (int s2 = 0; s2 < 2; ++s2) {
        const int d0 = s2 * 32;
        const int ab = (rh * 16 + lm) * QS_LD + dc + d0 + hl * 8;
        v16bf a = ld_frag(Qs, ab, ab + 16);
        const int bb = (cq * 16 + lm) * KV_LD + d0 + hl * 8;
        v16bf kb = ld_frag(KsC, bb, bb + 16);
        sacc = __builtin_amdgcn_wmma_f32_16x16x32_bf16(false, a, false, kb,
                                                       (short)0, sacc, false, false);
      }
    }

    // ============ epilogue: residual, new_prev, masks, softmax ============
    const int colg = k0 + cq * 16 + lm;
    const float pad_pen = mrow[colg] ? 0.f : PEN;
    if (kt + 1 < NKT)             // warm next prev tile (global_prefetch_b8)
      __builtin_prefetch(Pv + (bq + q0 + rh * 16 + lm) * (long)NSK + k0 + TBN, 0, 1);

    #pragma unroll
    for (int i = 0; i < 8; ++i) {
      const int rowg = q0 + rh * 16 + i + 8 * hl;
      const long pidx = (bq + rowg) * (long)NSK + colg;
      float e = sacc[i] * SCALE + Pv[pidx];
      Np[pidx] = e;               // new_prev is pre-mask
      sv[i] = e - ((colg > rowg) ? PEN : 0.f) - pad_pen;
    }

    if (active) {
      #pragma unroll
      for (int i = 0; i < 8; ++i) {      // row max over this wave's 16 columns
        float r = sv[i];
        r = fmaxf(r, __shfl_xor(r, 1, 32));
        r = fmaxf(r, __shfl_xor(r, 2, 32));
        r = fmaxf(r, __shfl_xor(r, 4, 32));
        r = fmaxf(r, __shfl_xor(r, 8, 32));
        tm[i] = r;
      }
      if (lm == 0) {
        #pragma unroll
        for (int i = 0; i < 8; ++i) s_red[cq * TBM + rh * 16 + 8 * hl + i] = tm[i];
      }
      __syncthreads();
      #pragma unroll
      for (int i = 0; i < 8; ++i) {
        const int rowl = rh * 16 + i + 8 * hl;
        float cm = fmaxf(fmaxf(s_red[0 * TBM + rowl], s_red[1 * TBM + rowl]),
                         fmaxf(s_red[2 * TBM + rowl], s_red[3 * TBM + rowl]));
        float nm = fmaxf(m_i[i], cm);
        corr[i] = __expf(m_i[i] - nm);
        m_i[i]  = nm;
        float p = __expf(sv[i] - nm);
        float rs = p;
        rs += __shfl_xor(rs, 1, 32);
        rs += __shfl_xor(rs, 2, 32);
        rs += __shfl_xor(rs, 4, 32);
        rs += __shfl_xor(rs, 8, 32);
        tm[i] = rs;
        Ps[rowl * PS_LD + cq * 16 + lm] = f2bf(p);
      }
      if (lm == 0) {
        #pragma unroll
        for (int i = 0; i < 8; ++i) s_red[128 + cq * TBM + rh * 16 + 8 * hl + i] = tm[i];
      }
      __syncthreads();
      #pragma unroll
      for (int i = 0; i < 8; ++i) {
        const int rowl = rh * 16 + i + 8 * hl;
        float ts = s_red[128 + 0 * TBM + rowl] + s_red[128 + 1 * TBM + rowl]
                 + s_red[128 + 2 * TBM + rowl] + s_red[128 + 3 * TBM + rowl];
        l_i[i] = l_i[i] * corr[i] + ts;
        #pragma unroll
        for (int t = 0; t < 8; ++t) o_acc[t][i] *= corr[i];
      }

      // ============ O += P V : 8 pipelined 64-wide d-chunks ============
      const int ar = (rh * 16 + lm) * PS_LD + hl * 8;
      v16bf a0 = ld_frag(Ps, ar +  0, ar + 16);   // keys 0..31
      v16bf a1 = ld_frag(Ps, ar + 32, ar + 48);   // keys 32..63
      #pragma unroll
      for (int c = 0; c < 8; ++c) {
        if (w == 0) tdm_wait0();
        __syncthreads();
        if (w == 0) {
          if (c < 7) {
            tdm_load_tile64(kv_b[(c + 1) & 1],
                            (unsigned long long)(size_t)(Vg + ((long)b * ND + (c + 1) * 64) * NSK + k0),
                            NSK);
          } else if (kt + 1 < NKT) {
            tdm_load_tile64(kv_b[0],
                            (unsigned long long)(size_t)(Kb + (bk + k0 + TBN) * ND), ND);
          }
        }
        const unsigned short* VtC = &KV2[c & 1][0];
        const int b0 = (cq * 16 + lm) * KV_LD + hl * 8;
        v16bf vb0 = ld_frag(VtC, b0,      b0 + 16);   // keys 0..31
        v16bf vb1 = ld_frag(VtC, b0 + 32, b0 + 48);   // keys 32..63
        o_acc[c] = __builtin_amdgcn_wmma_f32_16x16x32_bf16(false, a0, false, vb0,
                                                           (short)0, o_acc[c], false, false);
        o_acc[c] = __builtin_amdgcn_wmma_f32_16x16x32_bf16(false, a1, false, vb1,
                                                           (short)0, o_acc[c], false, false);
      }
    }
  }

  // ---- normalize and write O (coalesced per half-wave)
  #pragma unroll
  for (int i = 0; i < 8; ++i) {
    const float inv = 1.0f / l_i[i];
    const int rowg = q0 + rh * 16 + i + 8 * hl;
    #pragma unroll
    for (int c = 0; c < 8; ++c) {
      const int col = c * 64 + cq * 16 + lm;
      Ov[(bq + rowg) * ND + col] = o_acc[c][i] * inv;
    }
  }
}

extern "C" void kernel_launch(void* const* d_in, const int* in_sizes, int n_in,
                              void* d_out, int out_size, void* d_ws, size_t ws_size,
                              hipStream_t stream) {
  const float* Q  = (const float*)d_in[0];
  const float* K  = (const float*)d_in[1];
  const float* V  = (const float*)d_in[2];
  const float* Pv = (const float*)d_in[3];
  const unsigned char* Mk = (const unsigned char*)d_in[4];
  float* Ov = (float*)d_out;                          // [B,Sq,D]
  float* Np = Ov + (size_t)NB * NSQ * ND;             // [B,Sq,Sk]

  // workspace: bf16 Q, bf16 K (layout preserved), bf16 V transposed [B][D][Sk]
  unsigned short* Qb = (unsigned short*)d_ws;
  unsigned short* Kb = Qb + (size_t)NB * NSQ * ND;
  unsigned short* Vg = Kb + (size_t)NB * NSK * ND;

  hipLaunchKernelGGL(cvt_bf16_kernel, dim3(2048), dim3(256), 0, stream, Q, Qb);
  hipLaunchKernelGGL(cvt_bf16_kernel, dim3(2048), dim3(256), 0, stream, K, Kb);
  hipLaunchKernelGGL(transpose_v_kernel, dim3(NSK / 64, ND / 64, NB), dim3(256), 0, stream, V, Vg);
  hipLaunchKernelGGL(fa_resid_kernel, dim3(NSQ / TBM, NB), dim3(256), 0, stream,
                     Qb, Kb, Vg, Pv, Mk, Ov, Np);
}
